// MHSAModule_82703890251936
// MI455X (gfx1250) — matrix operand order repaired and checked by reading, not compiled, not run on hardware
//
#include <hip/hip_runtime.h>

// Transformer-XL MHSA forward for MI455X (gfx1250, wave32, WMMA).
// All matmuls in bf16 -> f32 accumulate via v_wmma_f32_16x16x32_bf16.
// Attention is fused flash-style; rel_shift handled analytically via an
// LDS-resident positional-score row band (17 rows x 1024 cols).
// K/V/P tiles are streamed into LDS with double-buffered
// GLOBAL_LOAD_ASYNC_TO_LDS_B128 (ASYNCcnt) for single-wave latency hiding.

typedef __bf16 bf16_t;
typedef __attribute__((ext_vector_type(16))) __bf16 v16bf;
typedef __attribute__((ext_vector_type(8)))  float  v8f;

#define D_MODEL 512
#define T_SEQ   1024
#define BATCH   8
#define HEADS   8
#define HDIM    64

__device__ __forceinline__ v8f wmma_bf16(v16bf a, v16bf b, v8f c) {
  // (neg_a, A, neg_b, B, c_mod, C, reuse_a, reuse_b)
  return __builtin_amdgcn_wmma_f32_16x16x32_bf16(false, a, false, b, (short)0, c,
                                                 false, false);
}

// ---- async global->LDS helpers (CDNA5 ASYNCcnt path) ------------------------
__device__ __forceinline__ unsigned lds_off32(const void* p) {
  // LDS aperture keeps the wave-relative byte offset in addr[31:0].
  return (unsigned)(unsigned long long)p;
}

__device__ __forceinline__ void async_copy_b128(unsigned lds_byte_off, const void* g) {
  asm volatile("global_load_async_to_lds_b128 %0, %1, off"
               :: "v"(lds_byte_off), "v"(g) : "memory");
}

__device__ __forceinline__ void wait_async_le4() {
  asm volatile("s_wait_asynccnt 0x4" ::: "memory");
}
__device__ __forceinline__ void wait_async_le16() {
  asm volatile("s_wait_asynccnt 0x10" ::: "memory");
}
__device__ __forceinline__ void wait_ds0() {
  asm volatile("s_wait_dscnt 0x0" ::: "memory");
}

// Stage BYTES contiguous bytes (multiple of 512) into LDS; BYTES/512 async ops.
template <int BYTES>
__device__ __forceinline__ void stage_linear(bf16_t* lds, const bf16_t* g, int lane) {
  const unsigned lbase = lds_off32(lds) + lane * 16;
  const char* gb = (const char*)g + lane * 16;
#pragma unroll
  for (int i = 0; i < BYTES / 512; ++i)
    async_copy_b128(lbase + i * 512, gb + i * 512);
}

// Stage a 64-row x 32-col bf16 tile of Vt (row stride T_SEQ elems) -> vbuf[64][32].
__device__ __forceinline__ void stage_vt(bf16_t* lds, const bf16_t* vmat, int j0, int lane) {
  const int rsub = lane >> 2;           // 0..7: row within group of 8
  const int cseg = (lane & 3) * 16;     // byte segment within 64B row
  const unsigned lbase = lds_off32(lds);
#pragma unroll
  for (int i = 0; i < 8; ++i) {
    const int f = i * 8 + rsub;
    async_copy_b128(lbase + f * 64 + cseg,
                    (const char*)(vmat + (size_t)f * T_SEQ + j0) + cseg);
  }
}

// A fragment: 16x32 (MxK) bf16 from row-major [rows, ld] storage.
// Contiguous k order (as B): lanes 0-15 hold rows M=0..15 with K = 0..15;
//             lanes 16-31 hold rows M=0..15 with K = 16..31.
__device__ __forceinline__ v16bf load_a_frag(const bf16_t* mat, int row0, int k0,
                                             int ld, int rowmax, int lane) {
  int row = row0 + (lane & 15);
  if (row > rowmax) row = rowmax;  // clamp (clamped rows are never consumed)
  const bf16_t* p = mat + (size_t)row * ld + k0 + ((lane < 16) ? 0 : 16);
  v16bf a;
#pragma unroll
  for (int e = 0; e < 8; ++e) a[e] = p[e];
#pragma unroll
  for (int e = 0; e < 8; ++e) a[8 + e] = p[8 + e];
  return a;
}

// B fragment: 32x16 (KxN) bf16 from [N, K]-major storage (row = n, K contiguous).
// Contiguous k: lanes 0-15 => n = lane, K = 0..15; lanes 16-31 => n = lane-16, K = 16..31.
__device__ __forceinline__ v16bf load_b_frag(const bf16_t* mat, int n0, int k0,
                                             int ld, int lane) {
  const bf16_t* p = mat + (size_t)(n0 + (lane & 15)) * ld + k0 + ((lane < 16) ? 0 : 16);
  v16bf b;
#pragma unroll
  for (int e = 0; e < 16; ++e) b[e] = p[e];
  return b;
}

// ---------------------------------------------------------------- prep kernels

// Cast+transpose a 512x512 f32 matrix into bf16 [C, R] ([N, K]-major for WMMA B).
__global__ void txl_tcast_kernel(const float* __restrict__ in, bf16_t* __restrict__ out) {
  int i = blockIdx.x * 256 + threadIdx.x;  // 512*512 elements
  int r = i >> 9, c = i & 511;
  out[(size_t)c * 512 + r] = (bf16_t)in[i];
}

__global__ void txl_cast_kernel(const float* __restrict__ in, bf16_t* __restrict__ out, int n) {
  int i = blockIdx.x * 256 + threadIdx.x;
  if (i < n) out[i] = (bf16_t)in[i];
}

// LayerNorm over D=512 per row; one wave per row; wave-level shfl reduction.
__global__ __launch_bounds__(32) void txl_ln_kernel(const float* __restrict__ x,
                                                    const float* __restrict__ gamma,
                                                    const float* __restrict__ beta,
                                                    bf16_t* __restrict__ xnb) {
  const int row = blockIdx.x;          // B*T rows
  const int lane = threadIdx.x & 31;
  const float* xr = x + (size_t)row * D_MODEL;
  float vals[16];
  float s = 0.f, ss = 0.f;
#pragma unroll
  for (int k = 0; k < 16; ++k) {
    float v_ = xr[lane + 32 * k];
    vals[k] = v_; s += v_; ss += v_ * v_;
  }
#pragma unroll
  for (int m = 1; m <= 16; m <<= 1) {
    s  += __shfl_xor(s,  m, 32);
    ss += __shfl_xor(ss, m, 32);
  }
  const float mu   = s * (1.f / D_MODEL);
  const float var  = ss * (1.f / D_MODEL) - mu * mu;
  const float rstd = rsqrtf(var + 1e-3f);
  bf16_t* o = xnb + (size_t)row * D_MODEL;
#pragma unroll
  for (int k = 0; k < 16; ++k) {
    int e = lane + 32 * k;
    o[e] = (bf16_t)((vals[k] - mu) * rstd * gamma[e] + beta[e]);
  }
}

// ------------------------------------------------------ QKVP projection (WMMA)
// mode 0: Q -> QU = q+bq+u, QV = q+bq+v  [B,H,T,d]
// mode 1: K -> Kb  [B,H,T,d]
// mode 2: V -> Vt  [B,H,d,T]  (transposed so P.V B-fragments are contiguous)
// mode 3: P -> Pb  [B,H,T,d]  (no bias)
__global__ __launch_bounds__(32) void txl_proj_kernel(
    const bf16_t* __restrict__ xnb, const bf16_t* __restrict__ posb,
    const bf16_t* __restrict__ wqT, const bf16_t* __restrict__ wkT,
    const bf16_t* __restrict__ wvT, const bf16_t* __restrict__ wpT,
    const float* __restrict__ bq, const float* __restrict__ bk,
    const float* __restrict__ bv, const float* __restrict__ u,
    const float* __restrict__ v,
    bf16_t* __restrict__ QU, bf16_t* __restrict__ QV, bf16_t* __restrict__ Kb,
    bf16_t* __restrict__ Vt, bf16_t* __restrict__ Pb) {
  const int lane = threadIdx.x & 31;
  const int mt = blockIdx.x;    // M tile (16 rows)
  const int nt = blockIdx.y;    // N tile (64 cols)
  const int mode = blockIdx.z;  // 0..3
  const bf16_t* A = (mode == 3) ? posb : xnb;
  const bf16_t* W = (mode == 0) ? wqT : (mode == 1) ? wkT : (mode == 2) ? wvT : wpT;
  const int m0 = mt * 16, n0 = nt * 64;
  v8f acc[4] = {};
  for (int k0 = 0; k0 < D_MODEL; k0 += 32) {
    v16bf a = load_a_frag(A, m0, k0, D_MODEL, BATCH * T_SEQ - 1, lane);
#pragma unroll
    for (int c = 0; c < 4; ++c) {
      v16bf bfr = load_b_frag(W, n0 + c * 16, k0, D_MODEL, lane);
      acc[c] = wmma_bf16(a, bfr, acc[c]);
    }
  }
  const int rbase = (lane < 16) ? 0 : 8;
  const int jl = lane & 15;
#pragma unroll
  for (int c = 0; c < 4; ++c) {
#pragma unroll
    for (int r = 0; r < 8; ++r) {
      const int m = m0 + rbase + r;
      const int n = n0 + c * 16 + jl;
      const int b_ = m >> 10, t = m & (T_SEQ - 1);
      const int h = n >> 6, f = n & (HDIM - 1);
      const float val = acc[c][r];
      const size_t obhtf = ((((size_t)b_ * HEADS + h) * T_SEQ + t) * HDIM + f);
      if (mode == 0) {
        float qb = val + bq[n];
        QU[obhtf] = (bf16_t)(qb + u[n]);
        QV[obhtf] = (bf16_t)(qb + v[n]);
      } else if (mode == 1) {
        Kb[obhtf] = (bf16_t)(val + bk[n]);
      } else if (mode == 2) {
        Vt[(((size_t)b_ * HEADS + h) * HDIM + f) * T_SEQ + t] = (bf16_t)(val + bv[n]);
      } else {
        Pb[obhtf] = (bf16_t)val;
      }
    }
  }
}

// --------------------------------------------------- fused attention (flash)
// One wave per (b, h, 16-query-row tile).
// rel_shift (S==T, no mask):
//   j <= i   : ps[i,   T-1-(i-j)]
//   j == i+1 : 0
//   j >  i+1 : ps[i+1, j-i-2]
__global__ __launch_bounds__(32) void txl_attn_kernel(
    const bf16_t* __restrict__ QU, const bf16_t* __restrict__ QV,
    const bf16_t* __restrict__ Kb, const bf16_t* __restrict__ Vt,
    const bf16_t* __restrict__ Pb, bf16_t* __restrict__ ctxb) {
  __shared__ bf16_t ps[17][T_SEQ];                 // positional score band
  __shared__ __align__(16) bf16_t pt[16][32];      // probability tile staging
  __shared__ __align__(16) bf16_t pbuf[2][16 * HDIM];  // P tile double buffer
  __shared__ __align__(16) bf16_t kbuf[2][32 * HDIM];  // K tile double buffer
  __shared__ __align__(16) bf16_t vbuf[2][HDIM * 32];  // Vt tile double buffer
  const int lane = threadIdx.x & 31;
  const int it = blockIdx.x;         // T/16 query tiles
  const int h = blockIdx.y;
  const int b = blockIdx.z;
  const int i0 = it * 16;
  const size_t bh = (size_t)b * HEADS + h;
  const bf16_t* qu   = QU + bh * T_SEQ * HDIM;
  const bf16_t* qv   = QV + bh * T_SEQ * HDIM;
  const bf16_t* kmat = Kb + bh * T_SEQ * HDIM;
  const bf16_t* vmat = Vt + bh * (size_t)HDIM * T_SEQ;
  const bf16_t* pmat = Pb + bh * T_SEQ * HDIM;
  const int rbase = (lane < 16) ? 0 : 8;
  const int jl = lane & 15;

  // phase 1: ps[r][m] = (q_{i0+r}+v) . p_m  for r in [0,16], all m.
  // P tiles (16x64 = 2KB, contiguous) double-buffered via async LDS copies.
  {
    v16bf a0[2], a1[2];
#pragma unroll
    for (int s = 0; s < 2; ++s) {
      a0[s] = load_a_frag(qv, i0,      32 * s, HDIM, T_SEQ - 1, lane);
      a1[s] = load_a_frag(qv, i0 + 16, 32 * s, HDIM, T_SEQ - 1, lane);  // only row 0 kept
    }
    stage_linear<2048>(pbuf[0], pmat, lane);
    for (int jt = 0; jt < T_SEQ / 16; ++jt) {
      const int cur = jt & 1, nxt = cur ^ 1;
      const int jn = (jt + 1 < T_SEQ / 16) ? jt + 1 : jt;  // clamp: redundant restage
      wait_ds0();  // prior DS reads of pbuf[nxt] complete (DS vs ASYNC unordered)
      stage_linear<2048>(pbuf[nxt], pmat + (size_t)jn * 16 * HDIM, lane);
      wait_async_le4();  // current tile resident
      v8f acc0 = {}; v8f acc1 = {};
#pragma unroll
      for (int s = 0; s < 2; ++s) {
        v16bf bb = load_b_frag(pbuf[cur], 0, 32 * s, HDIM, lane);
        acc0 = wmma_bf16(a0[s], bb, acc0);
        acc1 = wmma_bf16(a1[s], bb, acc1);
      }
      const int col = jt * 16 + jl;
#pragma unroll
      for (int r = 0; r < 8; ++r) ps[rbase + r][col] = (bf16_t)acc0[r];
      if (lane < 16) ps[16][col] = (bf16_t)acc1[0];  // row i0+16 (unused when i0+16>=T)
    }
  }
  __syncthreads();

  // phase 2: streaming (online-softmax) pass over key tiles of 32, with
  // K (4KB) and Vt (4KB) tiles double-buffered via async LDS copies.
  v16bf aq[2];
#pragma unroll
  for (int s = 0; s < 2; ++s) aq[s] = load_a_frag(qu, i0, 32 * s, HDIM, T_SEQ - 1, lane);
  float mrow[8], lrow[8];
  v8f oacc[4] = {};
#pragma unroll
  for (int r = 0; r < 8; ++r) { mrow[r] = -3.0e38f; lrow[r] = 0.f; }
  const float scale = 0.125f;  // 1/sqrt(64)

  stage_linear<4096>(kbuf[0], kmat, lane);
  stage_vt(vbuf[0], vmat, 0, lane);
  for (int jt = 0; jt < T_SEQ / 32; ++jt) {
    const int j0 = jt * 32;
    const int cur = jt & 1, nxt = cur ^ 1;
    const int j0n = (j0 + 32 < T_SEQ) ? j0 + 32 : j0;  // clamp: redundant restage
    wait_ds0();
    stage_linear<4096>(kbuf[nxt], kmat + (size_t)j0n * HDIM, lane);
    stage_vt(vbuf[nxt], vmat, j0n, lane);
    wait_async_le16();  // current K/V tiles resident (16 new ops in flight)

    v8f sc[2];
#pragma unroll
    for (int c = 0; c < 2; ++c) {
      v8f a_ = {};
#pragma unroll
      for (int s = 0; s < 2; ++s) {
        v16bf bb = load_b_frag(kbuf[cur], c * 16, 32 * s, HDIM, lane);
        a_ = wmma_bf16(aq[s], bb, a_);
      }
      sc[c] = a_;
    }
    // content + rel-shifted positional, scaled; per-row max via half-wave shuffles.
    float rm[8];
#pragma unroll
    for (int r = 0; r < 8; ++r) {
      const int i = i0 + rbase + r;
      float rmx = -3.0e38f;
#pragma unroll
      for (int c = 0; c < 2; ++c) {
        const int j = j0 + c * 16 + jl;
        float pv;
        if (j <= i)          pv = (float)ps[rbase + r][T_SEQ - 1 - (i - j)];
        else if (j == i + 1) pv = 0.f;
        else                 pv = (float)ps[rbase + r + 1][j - i - 2];
        const float s_ = (sc[c][r] + pv) * scale;
        sc[c][r] = s_;
        rmx = fmaxf(rmx, s_);
      }
#pragma unroll
      for (int m = 1; m <= 8; m <<= 1) rmx = fmaxf(rmx, __shfl_xor(rmx, m, 32));
      rm[r] = rmx;
    }
    __syncthreads();
#pragma unroll
    for (int r = 0; r < 8; ++r) {
      const float mnew = fmaxf(mrow[r], rm[r]);
      const float corr = __expf(mrow[r] - mnew);
      float psum = 0.f;
#pragma unroll
      for (int c = 0; c < 2; ++c) {
        const float p = __expf(sc[c][r] - mnew);
        psum += p;
        pt[rbase + r][c * 16 + jl] = (bf16_t)p;
      }
#pragma unroll
      for (int m = 1; m <= 8; m <<= 1) psum += __shfl_xor(psum, m, 32);
      lrow[r] = lrow[r] * corr + psum;
      mrow[r] = mnew;
#pragma unroll
      for (int fc = 0; fc < 4; ++fc) oacc[fc][r] *= corr;
    }
    __syncthreads();
    v16bf ap = load_a_frag(&pt[0][0], 0, 0, 32, 15, lane);
#pragma unroll
    for (int fc = 0; fc < 4; ++fc) {
      v16bf bb = load_b_frag(vbuf[cur], fc * 16, 0, 32, lane);  // vbuf is [f, t-local]
      oacc[fc] = wmma_bf16(ap, bb, oacc[fc]);
    }
  }
  // write context as [b, t, h, f] so it is the A operand of the output GEMM.
#pragma unroll
  for (int r = 0; r < 8; ++r) {
    const int t = i0 + rbase + r;
    const float rl = 1.f / lrow[r];
#pragma unroll
    for (int fc = 0; fc < 4; ++fc) {
      const int f = fc * 16 + jl;
      ctxb[(((size_t)b * T_SEQ + t) * HEADS + h) * HDIM + f] = (bf16_t)(oacc[fc][r] * rl);
    }
  }
}

// ------------------------------------------- output projection + residual add
__global__ __launch_bounds__(32) void txl_outproj_kernel(
    const bf16_t* __restrict__ ctxb, const bf16_t* __restrict__ woT,
    const float* __restrict__ bo, const float* __restrict__ inputs,
    float* __restrict__ out) {
  const int lane = threadIdx.x & 31;
  const int m0 = blockIdx.x * 16;
  const int n0 = blockIdx.y * 64;
  v8f acc[4] = {};
  for (int k0 = 0; k0 < D_MODEL; k0 += 32) {
    v16bf a = load_a_frag(ctxb, m0, k0, D_MODEL, BATCH * T_SEQ - 1, lane);
#pragma unroll
    for (int c = 0; c < 4; ++c) {
      v16bf bfr = load_b_frag(woT, n0 + c * 16, k0, D_MODEL, lane);
      acc[c] = wmma_bf16(a, bfr, acc[c]);
    }
  }
  const int rbase = (lane < 16) ? 0 : 8;
  const int jl = lane & 15;
#pragma unroll
  for (int c = 0; c < 4; ++c) {
#pragma unroll
    for (int r = 0; r < 8; ++r) {
      const size_t m = m0 + rbase + r;
      const int n = n0 + c * 16 + jl;
      out[m * D_MODEL + n] = acc[c][r] + bo[n] + inputs[m * D_MODEL + n];
    }
  }
}

// ----------------------------------------------------------------- launcher
extern "C" void kernel_launch(void* const* d_in, const int* in_sizes, int n_in,
                              void* d_out, int out_size, void* d_ws, size_t ws_size,
                              hipStream_t stream) {
  (void)in_sizes; (void)n_in; (void)out_size; (void)ws_size;
  const float* inputs = (const float*)d_in[0];
  const float* pos    = (const float*)d_in[1];
  const float* gamma  = (const float*)d_in[2];
  const float* beta   = (const float*)d_in[3];
  const float* wq     = (const float*)d_in[4];
  const float* bq     = (const float*)d_in[5];
  const float* wk     = (const float*)d_in[6];
  const float* bk     = (const float*)d_in[7];
  const float* wv     = (const float*)d_in[8];
  const float* bv     = (const float*)d_in[9];
  const float* wpos   = (const float*)d_in[10];
  const float* u      = (const float*)d_in[11];
  const float* v      = (const float*)d_in[12];
  const float* wo     = (const float*)d_in[13];
  const float* bo     = (const float*)d_in[14];
  float* out = (float*)d_out;

  char* ws = (char*)d_ws;
  size_t off = 0;
  auto take = [&](size_t bytes) -> char* {
    char* p = ws + off;
    off += (bytes + 255) & ~(size_t)255;
    return p;
  };
  const size_t MTOK = (size_t)BATCH * T_SEQ;                 // 8192 rows
  const size_t WMAT = (size_t)D_MODEL * D_MODEL * sizeof(bf16_t);
  const size_t BHTD = (size_t)BATCH * HEADS * T_SEQ * HDIM * sizeof(bf16_t);
  bf16_t* xnb  = (bf16_t*)take(MTOK * D_MODEL * sizeof(bf16_t));
  bf16_t* posb = (bf16_t*)take(MTOK * D_MODEL * sizeof(bf16_t));
  bf16_t* wqT  = (bf16_t*)take(WMAT);
  bf16_t* wkT  = (bf16_t*)take(WMAT);
  bf16_t* wvT  = (bf16_t*)take(WMAT);
  bf16_t* wpT  = (bf16_t*)take(WMAT);
  bf16_t* woT  = (bf16_t*)take(WMAT);
  bf16_t* QUb  = (bf16_t*)take(BHTD);
  bf16_t* QVb  = (bf16_t*)take(BHTD);
  bf16_t* Kbf  = (bf16_t*)take(BHTD);
  bf16_t* Vtb  = (bf16_t*)take(BHTD);
  bf16_t* Pbf  = (bf16_t*)take(BHTD);
  bf16_t* ctxb = (bf16_t*)take(MTOK * D_MODEL * sizeof(bf16_t));

  // weight transpose+cast (all [512,512] views)
  txl_tcast_kernel<<<1024, 256, 0, stream>>>(wq,   wqT);
  txl_tcast_kernel<<<1024, 256, 0, stream>>>(wk,   wkT);
  txl_tcast_kernel<<<1024, 256, 0, stream>>>(wv,   wvT);
  txl_tcast_kernel<<<1024, 256, 0, stream>>>(wpos, wpT);
  txl_tcast_kernel<<<1024, 256, 0, stream>>>(wo,   woT);
  // pre-LN + bf16 cast
  txl_ln_kernel<<<(int)MTOK, 32, 0, stream>>>(inputs, gamma, beta, xnb);
  txl_cast_kernel<<<(int)((MTOK * D_MODEL + 255) / 256), 256, 0, stream>>>(
      pos, posb, (int)(MTOK * D_MODEL));
  // Q/K/V/P projections
  txl_proj_kernel<<<dim3(512, 8, 4), 32, 0, stream>>>(
      xnb, posb, wqT, wkT, wvT, wpT, bq, bk, bv, u, v, QUb, QVb, Kbf, Vtb, Pbf);
  // fused relative attention
  txl_attn_kernel<<<dim3(T_SEQ / 16, HEADS, BATCH), 32, 0, stream>>>(
      QUb, QVb, Kbf, Vtb, Pbf, ctxb);
  // output projection + bias + residual
  txl_outproj_kernel<<<dim3(512, 8), 32, 0, stream>>>(ctxb, woT, bo, inputs, out);
}
